// word2vec_84920093376668
// MI455X (gfx1250) — matrix operand order repaired and checked by reading, not compiled, not run on hardware
//
#include <hip/hip_runtime.h>
#include <hip/hip_bf16.h>
#include <cmath>

#define VOCAB 50257
#define EMB   128
#define CTX   200

typedef __attribute__((ext_vector_type(16))) _Float16 v16h;
typedef __attribute__((ext_vector_type(8)))  float    v8f;

// ---------------- K1: s[v] = sum_t text[t,v]  (40.2 MB streamed, coalesced over v)
__global__ void k_colsum(const float* __restrict__ text, float* __restrict__ s) {
    int v = blockIdx.x * blockDim.x + threadIdx.x;
    if (v >= VOCAB) return;
    float acc = 0.0f;
    const float* p = text + v;
#pragma unroll 4
    for (int t = 0; t < CTX; ++t) acc += p[(size_t)t * VOCAB];
    s[v] = acc;
}

// ---------------- K2: hidden[e] = dot(Wq[e,:], s)/CTX + bq[e]   (one block per e, coalesced)
__global__ void k_hidden(const float* __restrict__ Wq, const float* __restrict__ bq,
                         const float* __restrict__ s, float* __restrict__ hidden) {
    __shared__ float red[256];
    const int e = blockIdx.x;
    const float* row = Wq + (size_t)e * VOCAB;
    float acc = 0.0f;
    for (int v = threadIdx.x; v < VOCAB; v += 256) acc += row[v] * s[v];
    red[threadIdx.x] = acc;
    __syncthreads();
    for (int off = 128; off > 0; off >>= 1) {
        if (threadIdx.x < (unsigned)off) red[threadIdx.x] += red[threadIdx.x + off];
        __syncthreads();
    }
    if (threadIdx.x == 0) hidden[e] = red[0] * (1.0f / (float)CTX) + bq[e];
}

// ---------------- K3: logits = Ww @ hidden + bw via v_wmma_f32_16x16x32_f16
// One wave per 16-row tile of Ww. B = hidden broadcast to all 16 columns, so D's
// columns are identical; lanes 0/16 hold rows m0..m0+7 / m0+8..m0+15 in C VGPRs 0..7.
__global__ void k_logits_wmma(const float* __restrict__ Ww, const float* __restrict__ bw,
                              const float* __restrict__ hidden, float* __restrict__ logits) {
    const int NT = (VOCAB + 15) / 16;                      // 3142 tiles
    const int wave = blockIdx.x * (blockDim.x >> 5) + (threadIdx.x >> 5);
    if (wave >= NT) return;                                // wave-uniform: EXEC stays all-1s
    const int lane = threadIdx.x & 31;
    const int m0   = wave * 16;
    int row = m0 + (lane & 15);
    if (row >= VOCAB) row = VOCAB - 1;                     // clamp (no divergence, just cndmask)
    const float* wrow  = Ww + (size_t)row * EMB;
    const int    ahalf = (lane < 16) ? 0 : 8;              // A layout: 16-bit 16x32 (ISA 7.12.2)
    const int    bhalf = (lane < 16) ? 0 : 8;              // B layout: same K order as A (K=bhalf..+7, bhalf+16..+23)

    v8f c = {};
#pragma unroll
    for (int k = 0; k < EMB; k += 32) {
        v16h a;
        const float* pa0 = wrow + k + ahalf;               // K = k+ahalf .. +7
        const float* pa1 = pa0 + 16;                       // K = k+ahalf+16 .. +23
#pragma unroll
        for (int j = 0; j < 8; ++j) {
            a[j]     = (_Float16)pa0[j];
            a[8 + j] = (_Float16)pa1[j];
        }
        v16h b;
        const float* pb = hidden + k + bhalf;              // broadcast within half-wave (L0 hits)
#pragma unroll
        for (int j = 0; j < 8; ++j) { b[j] = (_Float16)pb[j]; b[8 + j] = (_Float16)pb[16 + j]; }

        c = __builtin_amdgcn_wmma_f32_16x16x32_f16(
                /*neg_a=*/false, a, /*neg_b=*/false, b,
                /*c_mod=*/(short)0, c, /*reuse_a=*/false, /*reuse_b=*/false);
    }

    // Column N=0 lives in lane 0 (M=r) and lane 16 (M=r+8), VGPR r = c[r]
    if (lane == 0 || lane == 16) {
        const int rbase = m0 + ((lane == 16) ? 8 : 0);
#pragma unroll
        for (int r = 0; r < 8; ++r) {
            const int vr = rbase + r;
            if (vr < VOCAB) logits[vr] = c[r] + bw[vr];
        }
    }
}

// ---------------- K4: single-block streaming logsumexp (online (m,s) merge)
__global__ void k_lse(const float* __restrict__ logits, float* __restrict__ lse) {
    __shared__ float sm[1024];
    __shared__ float ss[1024];
    const int tid = threadIdx.x;
    float m = -INFINITY, s = 0.0f;
    for (int i = tid; i < VOCAB; i += 1024) {
        const float x = logits[i];
        if (x > m) { s = s * expf(m - x) + 1.0f; m = x; }
        else       { s += expf(x - m); }
    }
    sm[tid] = m; ss[tid] = s;
    __syncthreads();
    for (int off = 512; off > 0; off >>= 1) {
        if (tid < off) {
            const float m1 = sm[tid], s1 = ss[tid];
            const float m2 = sm[tid + off], s2 = ss[tid + off];
            const float M = fmaxf(m1, m2);
            ss[tid] = s1 * expf(m1 - M) + s2 * expf(m2 - M);
            sm[tid] = M;
        }
        __syncthreads();
    }
    if (tid == 0) lse[0] = sm[0] + logf(ss[0]);
}

// ---------------- K5: out = logits - lse
__global__ void k_logsoftmax(const float* __restrict__ logits, const float* __restrict__ lse,
                             float* __restrict__ out) {
    const int v = blockIdx.x * blockDim.x + threadIdx.x;
    if (v < VOCAB) out[v] = logits[v] - lse[0];
}

extern "C" void kernel_launch(void* const* d_in, const int* in_sizes, int n_in,
                              void* d_out, int out_size, void* d_ws, size_t ws_size,
                              hipStream_t stream) {
    (void)in_sizes; (void)n_in; (void)out_size; (void)ws_size;
    const float* text = (const float*)d_in[0];
    // d_in[1] = length (int scalar) == CTX, baked in at compile time
    const float* Wq   = (const float*)d_in[2];
    const float* bq   = (const float*)d_in[3];
    const float* Ww   = (const float*)d_in[4];
    const float* bw   = (const float*)d_in[5];
    float* out = (float*)d_out;

    // workspace layout (floats): s[50304 pad] | hidden[128] | logits[50304 pad] | lse[1]
    float* ws     = (float*)d_ws;
    float* s      = ws;
    float* hidden = ws + 50304;
    float* logits = ws + 50304 + 128;
    float* lse    = ws + 50304 + 128 + 50304;

    k_colsum<<<(VOCAB + 255) / 256, 256, 0, stream>>>(text, s);
    k_hidden<<<EMB, 256, 0, stream>>>(Wq, bq, s, hidden);

    const int NT = (VOCAB + 15) / 16;              // 3142 wave-tiles
    const int wavesPerBlock = 8;                   // 256 threads = 8 wave32s
    const int blocks = (NT + wavesPerBlock - 1) / wavesPerBlock;
    k_logits_wmma<<<blocks, 32 * wavesPerBlock, 0, stream>>>(Ww, bw, hidden, logits);

    k_lse<<<1, 1024, 0, stream>>>(logits, lse);
    k_logsoftmax<<<(VOCAB + 255) / 256, 256, 0, stream>>>(logits, lse, out);
}
